// TemAttentionLayer_74302934221288
// MI455X (gfx1250) — compile-verified, matrix operand order checked by reading, not run on hardware
//
#include <hip/hip_runtime.h>
#include <hip/hip_bf16.h>

typedef __bf16 bf16_t;
typedef __attribute__((ext_vector_type(16))) __bf16 v16bf;
typedef __attribute__((ext_vector_type(8)))  __bf16 v8bf;
typedef __attribute__((ext_vector_type(8)))  float  v8f;

constexpr int B_ = 16, N_ = 64, T_ = 128, D_ = 256, H_ = 8, HD_ = 32, K_ = 3;

__device__ __forceinline__ v8f wmma_bf16(v16bf a, v16bf b, v8f c) {
  // D(f32 16x16) = A(bf16 16x32) * B(bf16 32x16) + C
  return __builtin_amdgcn_wmma_f32_16x16x32_bf16(false, a, false, b, (short)0, c,
                                                 false, false);
}

// Load one 16x32 bf16 WMMA operand fragment from a row-major [rows][stride]
// array. p points at element [row0][k0]. CDNA5 16-bit A/B layout:
//   lanes 0-15 : row = lane,    K = {k0..k0+7, k0+16..k0+23}
//   lanes 16-31: row = lane-16, K = {k0+8..k0+15, k0+24..k0+31}
__device__ __forceinline__ v16bf load_frag(const bf16_t* p, int stride, int lane) {
  const bf16_t* q = p + (lane & 15) * stride + ((lane & 16) ? 8 : 0);
  v8bf lo = *(const v8bf*)q;
  v8bf hi = *(const v8bf*)(q + 16);
  v16bf r;
#pragma unroll
  for (int i = 0; i < 8; ++i) { r[i] = lo[i]; r[i + 8] = hi[i]; }
  return r;
}

// ---------------- stage 0: weight conversion / repack ----------------------

// Wq/Wk: f32 [o][i][1][k]  ->  bf16 [k][o][i]  (per-tap row-major GEMM-B)
__global__ void k_cvt_convw(const float* __restrict__ W, bf16_t* __restrict__ dst) {
  int idx = blockIdx.x * blockDim.x + threadIdx.x;
  if (idx >= D_ * D_ * K_) return;
  int k = idx % K_;
  int i = (idx / K_) % D_;
  int o = idx / (K_ * D_);
  dst[(k * D_ + o) * D_ + i] = (bf16_t)W[idx];
}

__global__ void k_cvt_densew(const float* __restrict__ W, bf16_t* __restrict__ dst,
                             int n) {
  int idx = blockIdx.x * blockDim.x + threadIdx.x;
  if (idx < n) dst[idx] = (bf16_t)W[idx];
}

// ---------------- stage 1: conv-QK + linear-V projections ------------------
// grid: (B*N, T/32), block 256 (8 waves). Each WG computes 32 time rows x 256
// features of q,k,v in bf16. Conv handled as 3 shifted-window GEMMs.
__global__ __launch_bounds__(256) void k_qkv(
    const float* __restrict__ query, const float* __restrict__ key,
    const float* __restrict__ value,
    const bf16_t* __restrict__ wq, const bf16_t* __restrict__ wk,
    const bf16_t* __restrict__ wv,
    const float* __restrict__ bq, const float* __restrict__ bk,
    const float* __restrict__ bv,
    bf16_t* __restrict__ qo, bf16_t* __restrict__ ko, bf16_t* __restrict__ vo) {
  __shared__ __attribute__((aligned(16))) bf16_t Xq[34 * D_];
  __shared__ __attribute__((aligned(16))) bf16_t Xk[34 * D_];
  __shared__ __attribute__((aligned(16))) bf16_t Xv[32 * D_];

  const int bn  = blockIdx.x;
  const int t0  = blockIdx.y * 32;
  const int tid = threadIdx.x;
  const size_t base = (size_t)bn * T_ * D_;

  // stage padded activation windows (f32 -> bf16)
  for (int idx = tid; idx < 34 * D_; idx += 256) {
    int r = idx >> 8, c = idx & (D_ - 1);
    int t = t0 - 1 + r;                       // window covers t0-1 .. t0+32
    bool in = (t >= 0) && (t < T_);
    float xq = in ? query[base + (size_t)t * D_ + c] : 0.0f;
    float xk = in ? key[base + (size_t)t * D_ + c] : 0.0f;
    Xq[idx] = (bf16_t)xq;
    Xk[idx] = (bf16_t)xk;
  }
  for (int idx = tid; idx < 32 * D_; idx += 256) {
    int r = idx >> 8, c = idx & (D_ - 1);
    Xv[idx] = (bf16_t)value[base + (size_t)(t0 + r) * D_ + c];
  }
  __syncthreads();

  const int wave = tid >> 5, lane = tid & 31;
  const int col = lane & 15;
  const int rowoff = (lane & 16) ? 8 : 0;

#pragma unroll
  for (int j = 0; j < 2; ++j) {
    const int n0 = (wave * 2 + j) * 16;       // output-feature tile base

    // ---- q: conv over time (3 taps, K-dim 768) ----
    {
      v8f acc[2] = {};
#pragma unroll
      for (int tap = 0; tap < 3; ++tap) {
        const bf16_t* W = wq + ((size_t)tap * D_ + n0) * D_;
#pragma unroll
        for (int kk = 0; kk < 8; ++kk) {
          v16bf bfrag = load_frag(W + kk * 32, D_, lane);
#pragma unroll
          for (int rt = 0; rt < 2; ++rt) {
            v16bf afrag = load_frag(Xq + (rt * 16 + tap) * D_ + kk * 32, D_, lane);
            acc[rt] = wmma_bf16(afrag, bfrag, acc[rt]);
          }
        }
      }
      float bias = bq[n0 + col];
#pragma unroll
      for (int rt = 0; rt < 2; ++rt)
#pragma unroll
        for (int i = 0; i < 8; ++i) {
          int t = t0 + rt * 16 + i + rowoff;
          qo[base + (size_t)t * D_ + n0 + col] = (bf16_t)(acc[rt][i] + bias);
        }
    }

    // ---- k: same shape ----
    {
      v8f acc[2] = {};
#pragma unroll
      for (int tap = 0; tap < 3; ++tap) {
        const bf16_t* W = wk + ((size_t)tap * D_ + n0) * D_;
#pragma unroll
        for (int kk = 0; kk < 8; ++kk) {
          v16bf bfrag = load_frag(W + kk * 32, D_, lane);
#pragma unroll
          for (int rt = 0; rt < 2; ++rt) {
            v16bf afrag = load_frag(Xk + (rt * 16 + tap) * D_ + kk * 32, D_, lane);
            acc[rt] = wmma_bf16(afrag, bfrag, acc[rt]);
          }
        }
      }
      float bias = bk[n0 + col];
#pragma unroll
      for (int rt = 0; rt < 2; ++rt)
#pragma unroll
        for (int i = 0; i < 8; ++i) {
          int t = t0 + rt * 16 + i + rowoff;
          ko[base + (size_t)t * D_ + n0 + col] = (bf16_t)(acc[rt][i] + bias);
        }
    }

    // ---- v: plain linear (K-dim 256) ----
    {
      v8f acc[2] = {};
#pragma unroll
      for (int kk = 0; kk < 8; ++kk) {
        v16bf bfrag = load_frag(wv + (size_t)n0 * D_ + kk * 32, D_, lane);
#pragma unroll
        for (int rt = 0; rt < 2; ++rt) {
          v16bf afrag = load_frag(Xv + rt * 16 * D_ + kk * 32, D_, lane);
          acc[rt] = wmma_bf16(afrag, bfrag, acc[rt]);
        }
      }
      float bias = bv[n0 + col];
#pragma unroll
      for (int rt = 0; rt < 2; ++rt)
#pragma unroll
        for (int i = 0; i < 8; ++i) {
          int t = t0 + rt * 16 + i + rowoff;
          vo[base + (size_t)t * D_ + n0 + col] = (bf16_t)(acc[rt][i] + bias);
        }
    }
  }
}

// ---------------- stage 2: per-(b,n,head) attention ------------------------
// grid: B*N*H, block 128 (4 waves). scores = q@k^T (one WMMA per 16x16 tile,
// K==HD==32), row softmax, out = P@V. P round-trips LDS (C-layout->A-layout).
__global__ __launch_bounds__(128) void k_attn(
    const bf16_t* __restrict__ qi, const bf16_t* __restrict__ ki,
    const bf16_t* __restrict__ vi, bf16_t* __restrict__ ao) {
  __shared__ __attribute__((aligned(16))) bf16_t Ks[T_ * HD_];  // [t][hd]
  __shared__ __attribute__((aligned(16))) bf16_t Vt[HD_ * T_];  // [hd][t]
  __shared__ __attribute__((aligned(16))) bf16_t Ps[T_ * T_];   // probabilities

  const int bnh = blockIdx.x;
  const int h = bnh & (H_ - 1);
  const int bn = bnh >> 3;
  const size_t base = (size_t)bn * T_ * D_ + h * HD_;
  const int tid = threadIdx.x;

  for (int idx = tid; idx < T_ * HD_ / 4; idx += 128) {
    int t = idx >> 3, c4 = idx & 7;
    *(unsigned long long*)&Ks[t * HD_ + c4 * 4] =
        *(const unsigned long long*)&ki[base + (size_t)t * D_ + c4 * 4];
  }
  for (int idx = tid; idx < T_ * HD_; idx += 128) {
    int t = idx >> 5, c = idx & 31;
    Vt[c * T_ + t] = vi[base + (size_t)t * D_ + c];   // transpose V
  }
  __syncthreads();

  const int wave = tid >> 5, lane = tid & 31;
  const int col = lane & 15;
  const int rowoff = (lane & 16) ? 8 : 0;
  const float scale = 0.17677669529663687f;           // 1/sqrt(32)

#pragma unroll
  for (int rr = 0; rr < 2; ++rr) {
    const int rt = wave * 2 + rr;                     // q row-tile 0..7

    // scores: one WMMA per 16x16 tile (K = HD = 32)
    v16bf aq = load_frag(qi + base + (size_t)(rt * 16) * D_, D_, lane);
    v8f sc[8];
#pragma unroll
    for (int ct = 0; ct < 8; ++ct) {
      v8f z = {};
      sc[ct] = wmma_bf16(aq, load_frag(Ks + ct * 16 * HD_, HD_, lane), z);
    }

    // softmax across 128 columns: 8 in-lane tiles + 16-lane half reduction
#pragma unroll
    for (int jj = 0; jj < 8; ++jj) {
      float m = -3.0e38f;
#pragma unroll
      for (int ct = 0; ct < 8; ++ct) m = fmaxf(m, sc[ct][jj]);
      m = fmaxf(m, __shfl_xor(m, 1, 32));
      m = fmaxf(m, __shfl_xor(m, 2, 32));
      m = fmaxf(m, __shfl_xor(m, 4, 32));
      m = fmaxf(m, __shfl_xor(m, 8, 32));
      float s = 0.0f;
#pragma unroll
      for (int ct = 0; ct < 8; ++ct) {
        float e = __expf((sc[ct][jj] - m) * scale);
        sc[ct][jj] = e;
        s += e;
      }
      s += __shfl_xor(s, 1, 32);
      s += __shfl_xor(s, 2, 32);
      s += __shfl_xor(s, 4, 32);
      s += __shfl_xor(s, 8, 32);
      float rinv = 1.0f / s;
#pragma unroll
      for (int ct = 0; ct < 8; ++ct) sc[ct][jj] *= rinv;
    }

    // spill P (C-layout regs) to LDS as bf16 so it can re-enter as A-layout
#pragma unroll
    for (int ct = 0; ct < 8; ++ct)
#pragma unroll
      for (int jj = 0; jj < 8; ++jj)
        Ps[(rt * 16 + jj + rowoff) * T_ + ct * 16 + col] = (bf16_t)sc[ct][jj];

    __syncthreads();   // uniform: every wave runs rr = 0,1

    // out tile [16 x 32] = P[16 x 128] @ V[128 x 32]
    v8f o[2] = {};
#pragma unroll
    for (int kk = 0; kk < 4; ++kk) {
      v16bf ap = load_frag(Ps + rt * 16 * T_ + kk * 32, T_, lane);
#pragma unroll
      for (int nt = 0; nt < 2; ++nt) {
        v16bf bv = load_frag(Vt + nt * 16 * T_ + kk * 32, T_, lane);
        o[nt] = wmma_bf16(ap, bv, o[nt]);
      }
    }
#pragma unroll
    for (int nt = 0; nt < 2; ++nt)
#pragma unroll
      for (int i = 0; i < 8; ++i) {
        int t = rt * 16 + i + rowoff;
        ao[base + (size_t)t * D_ + nt * 16 + col] = (bf16_t)o[nt][i];
      }
  }
}

// ---------------- stage 3: output projection (fp32 out) --------------------
// grid: (B*N*T)/64, block 256. out = attn_out @ Wo^T + bo.
__global__ __launch_bounds__(256) void k_outproj(
    const bf16_t* __restrict__ xi, const bf16_t* __restrict__ wo,
    const float* __restrict__ bo, float* __restrict__ out) {
  __shared__ __attribute__((aligned(16))) bf16_t As[64 * D_];
  const int tid = threadIdx.x;
  const size_t row0 = (size_t)blockIdx.x * 64;

  const bf16_t* src = xi + row0 * D_;       // 64 rows are contiguous
  for (int idx = tid; idx < 64 * D_ / 8; idx += 256)
    *(uint4*)&As[idx * 8] = *(const uint4*)&src[idx * 8];
  __syncthreads();

  const int wave = tid >> 5, lane = tid & 31;
  const int col = lane & 15;
  const int rowoff = (lane & 16) ? 8 : 0;

#pragma unroll
  for (int j = 0; j < 2; ++j) {
    const int n0 = (wave * 2 + j) * 16;
    v8f acc[4] = {};
#pragma unroll
    for (int kk = 0; kk < 8; ++kk) {
      if (kk + 1 < 8)
        __builtin_prefetch((const void*)(wo + (size_t)n0 * D_ + (kk + 1) * 32), 0, 1);
      v16bf bw = load_frag(wo + (size_t)n0 * D_ + kk * 32, D_, lane);
#pragma unroll
      for (int rt = 0; rt < 4; ++rt) {
        v16bf aa = load_frag(As + rt * 16 * D_ + kk * 32, D_, lane);
        acc[rt] = wmma_bf16(aa, bw, acc[rt]);
      }
    }
    float bias = bo[n0 + col];
#pragma unroll
    for (int rt = 0; rt < 4; ++rt)
#pragma unroll
      for (int i = 0; i < 8; ++i)
        out[(row0 + rt * 16 + i + rowoff) * D_ + n0 + col] = acc[rt][i] + bias;
  }
}

// ---------------------------------------------------------------------------

extern "C" void kernel_launch(void* const* d_in, const int* in_sizes, int n_in,
                              void* d_out, int out_size, void* d_ws, size_t ws_size,
                              hipStream_t stream) {
  (void)in_sizes; (void)n_in; (void)out_size; (void)ws_size;
  const float* query = (const float*)d_in[0];
  const float* key   = (const float*)d_in[1];
  const float* value = (const float*)d_in[2];
  const float* Wq = (const float*)d_in[3];
  const float* bq = (const float*)d_in[4];
  const float* Wk = (const float*)d_in[5];
  const float* bk = (const float*)d_in[6];
  const float* Wv = (const float*)d_in[7];
  const float* bv = (const float*)d_in[8];
  const float* Wo = (const float*)d_in[9];
  const float* bo = (const float*)d_in[10];

  char* ws = (char*)d_ws;
  size_t off = 0;
  auto take = [&](size_t bytes) -> void* {
    void* p = (void*)(ws + off);
    off += (bytes + 255) & ~(size_t)255;
    return p;
  };
  bf16_t* wq_b = (bf16_t*)take((size_t)K_ * D_ * D_ * 2);
  bf16_t* wk_b = (bf16_t*)take((size_t)K_ * D_ * D_ * 2);
  bf16_t* wv_b = (bf16_t*)take((size_t)D_ * D_ * 2);
  bf16_t* wo_b = (bf16_t*)take((size_t)D_ * D_ * 2);
  const size_t act = (size_t)B_ * N_ * T_ * D_;
  bf16_t* q_b = (bf16_t*)take(act * 2);
  bf16_t* k_b = (bf16_t*)take(act * 2);
  bf16_t* v_b = (bf16_t*)take(act * 2);
  bf16_t* a_b = (bf16_t*)take(act * 2);

  k_cvt_convw<<<(D_ * D_ * K_ + 255) / 256, 256, 0, stream>>>(Wq, wq_b);
  k_cvt_convw<<<(D_ * D_ * K_ + 255) / 256, 256, 0, stream>>>(Wk, wk_b);
  k_cvt_densew<<<(D_ * D_ + 255) / 256, 256, 0, stream>>>(Wv, wv_b, D_ * D_);
  k_cvt_densew<<<(D_ * D_ + 255) / 256, 256, 0, stream>>>(Wo, wo_b, D_ * D_);

  k_qkv<<<dim3(B_ * N_, T_ / 32), 256, 0, stream>>>(
      query, key, value, wq_b, wk_b, wv_b, bq, bk, bv, q_b, k_b, v_b);

  k_attn<<<B_ * N_ * H_, 128, 0, stream>>>(q_b, k_b, v_b, a_b);

  k_outproj<<<(B_ * N_ * T_) / 64, 256, 0, stream>>>(a_b, wo_b, bo, (float*)d_out);
}
